// PrototypeRoutedLinear_82729660056157
// MI455X (gfx1250) — compile-verified, matrix operand order checked by reading, not run on hardware
//
#include <hip/hip_runtime.h>
#include <hip/hip_bf16.h>

#define IN_DIM  1024
#define OUT_DIM 1024
#define N_PROTO 32
#define RANK    16

typedef __attribute__((ext_vector_type(16))) __bf16 v16bf;
typedef __attribute__((ext_vector_type(8)))  float  v8f;
typedef __attribute__((ext_vector_type(4)))  float  v4f;
typedef __attribute__((ext_vector_type(2)))  float  v2f;

union FragB { v16bf v; unsigned int u[8]; };

__device__ __forceinline__ unsigned short f2bf(float f) {
    union { __bf16 h; unsigned short s; } u;
    u.h = (__bf16)f;
    return u.s;
}

// ---------------------------------------------------------------------------
// Prep kernel 1: convert B (32,16,1024) and A (32,1024,16) f32 -> bf16 in ws.
// ---------------------------------------------------------------------------
__global__ void prl_convert(const float* __restrict__ Bw, const float* __restrict__ Aw,
                            unsigned short* __restrict__ Bh, unsigned short* __restrict__ Ah) {
    const int nB = N_PROTO * RANK * IN_DIM;
    const int nA = N_PROTO * OUT_DIM * RANK;
    int stride = gridDim.x * blockDim.x;
    for (int i = blockIdx.x * blockDim.x + threadIdx.x; i < nB; i += stride)
        Bh[i] = f2bf(Bw[i]);
    for (int i = blockIdx.x * blockDim.x + threadIdx.x; i < nA; i += stride)
        Ah[i] = f2bf(Aw[i]);
}

// ---------------------------------------------------------------------------
// Prep kernel 2: ||proto||^2 per prototype (32 lanes, one proto each).
// ---------------------------------------------------------------------------
__global__ void prl_protosq(const float* __restrict__ P, float* __restrict__ p2) {
    int e = threadIdx.x;  // 32 threads
    const v4f* p = (const v4f*)(P + (size_t)e * IN_DIM);
    float s = 0.f;
    for (int i = 0; i < IN_DIM / 4; ++i) {
        v4f v = p[i];
        s += v.x * v.x + v.y * v.y + v.z * v.z + v.w * v.w;
    }
    p2[e] = s;
}

// ---------------------------------------------------------------------------
// Main fused kernel: 2 waves (64 threads) per 16-token tile.
//   Stage 0a: X tile staged into LDS (f32) via GLOBAL_LOAD_ASYNC_TO_LDS_B128
//   Stage 0b: ||x||^2 (ds_add_f32) + bf16 conversion, both from LDS
//   Stage 1 : distances via V_WMMA_F32_16X16X4_F32 (wave w -> protos w*16..)
//   Stage 1b: top-2 + renormalized softmax weights
//   Stage 2a: per active expert (split odd/even across waves),
//             h = X @ B[e]^T via V_WMMA_F32_16X16X32_BF16, weight-folded
//   Stage 2b: per 16-wide out tile (split odd/even), one bf16 WMMA per expert,
//             K pad slot 16 carries w (A) x bias (B); store f32 out.
// ---------------------------------------------------------------------------
__global__ __launch_bounds__(64) void prl_main(
    const float* __restrict__ x, const float* __restrict__ P,
    const float* __restrict__ bias, const float* __restrict__ temp,
    const unsigned short* __restrict__ Bh, const unsigned short* __restrict__ Ah,
    const float* __restrict__ p2g, float* __restrict__ out)
{
    __shared__ __align__(16) float          xtf[16 * IN_DIM];         // 64 KB f32 X tile
    __shared__ __align__(16) unsigned short xt [16 * IN_DIM];         // 32 KB bf16 X tile
    __shared__ __align__(16) unsigned short hbuf[N_PROTO * 16 * 16];  // 16 KB weighted h
    __shared__ float dls[16 * N_PROTO];
    __shared__ float x2s[16];
    __shared__ float wvs[N_PROTO * 16];
    __shared__ int   actE[N_PROTO];
    __shared__ unsigned int maskLds;
    __shared__ int   nActS;

    const int tid  = threadIdx.x;      // 0..63
    const int w    = tid >> 5;         // wave id (0,1) -- wave uniform
    const int lane = tid & 31;
    const int hi   = lane >> 4;        // half-wave id
    const int m    = lane & 15;        // row / column within 16 (per wave)
    const int mt   = tid & 15;         // token for 64-thread stages
    const int q    = tid >> 4;         // quarter-row 0..3
    const int t0   = blockIdx.x * 16;

    if (tid == 0) maskLds = 0u;
    if (tid < 16) x2s[tid] = 0.f;

    // ---- Stage 0a: async-stage X tile (f32) into LDS ----
    {
        const float* gsrc = x + (size_t)(t0 + mt) * IN_DIM + q * 256;
        float*       ldst = &xtf[mt * IN_DIM + q * 256];
        for (int i = 0; i < 256; i += 4) {
            unsigned loff = (unsigned)(uintptr_t)(ldst + i);
            const float* gp = gsrc + i;
            asm volatile("global_load_async_to_lds_b128 %0, %1, off"
                         :: "v"(loff), "v"(gp) : "memory");
        }
        asm volatile("s_wait_asynccnt 0x0" ::: "memory");
    }
    __syncthreads();

    // ---- Stage 0b: ||x||^2 partials + bf16 conversion (LDS -> LDS) ----
    {
        const float*    srcp = &xtf[mt * IN_DIM + q * 256];
        unsigned short* dstp = &xt [mt * IN_DIM + q * 256];
        float acc = 0.f;
        for (int i = 0; i < 256; i += 4) {
            v4f v = *(const v4f*)(srcp + i);
            acc += v.x * v.x + v.y * v.y + v.z * v.z + v.w * v.w;
            dstp[i + 0] = f2bf(v.x);
            dstp[i + 1] = f2bf(v.y);
            dstp[i + 2] = f2bf(v.z);
            dstp[i + 3] = f2bf(v.w);
        }
        atomicAdd(&x2s[mt], acc);   // ds_add_f32
    }
    __syncthreads();

    // ---- Stage 1: routing GEMM, wave w covers protos [16w, 16w+16) ----
    const int pbase = w * 16;
    {
        v8f sacc = {};
        const float* aptr = &xtf[m * IN_DIM + 2 * hi];                 // LDS A-frags
        const float* brow = P + (size_t)(pbase + m) * IN_DIM + 2 * hi; // proto col
        for (int k = 0; k < IN_DIM; k += 4) {
            v2f af = *(const v2f*)(aptr + k);
            v2f bf = *(const v2f*)(brow + k);
            sacc = __builtin_amdgcn_wmma_f32_16x16x4_f32(false, af, false, bf,
                                                         (short)0, sacc, false, false);
        }
        float p2v = p2g[pbase + m];
        for (int j = 0; j < 8; ++j) {
            int row = hi * 8 + j;
            float d2 = x2s[row] + p2v - 2.f * sacc[j];
            dls[row * N_PROTO + pbase + m] = sqrtf(fmaxf(d2, 0.f));
        }
    }
    __syncthreads();

    // ---- Stage 1b: top-2 + renormalized softmax weights (token = tid&15) ----
    float w0, w1; int i0, i1;
    {
        float d0 = 1e30f, d1 = 1e30f; i0 = 0; i1 = 0;
        for (int e = 0; e < N_PROTO; ++e) {
            float d = dls[mt * N_PROTO + e];
            if (d < d0)      { d1 = d0; i1 = i0; d0 = d; i0 = e; }
            else if (d < d1) { d1 = d;  i1 = e; }
        }
        float tcl = fmaxf(fabsf(temp[0]), 0.1f);
        // softmax renormalized over top-2: global denominator cancels
        float z = expf((d0 - d1) / tcl);    // = exp(s1 - s0) <= 1
        w0 = 1.f / (1.f + z);
        w1 = 1.f - w0;
        atomicOr(&maskLds, (1u << i0) | (1u << i1));
    }
    __syncthreads();
    if (tid == 0) {
        unsigned int mk = maskLds; int c = 0;
        while (mk) { int e = __ffs(mk) - 1; actE[c++] = e; mk &= mk - 1; }
        nActS = c;
    }
    __syncthreads();
    const int nAct = nActS;                 // block-uniform
    for (int s = 0; s < nAct; ++s) {
        int e = actE[s];
        wvs[s * 16 + mt] = (i0 == e) ? w0 : ((i1 == e) ? w1 : 0.f);
    }
    __syncthreads();

    // ---- Stage 2a: h[s] = (X @ B[e]^T) * w, experts split across waves ----
    for (int s = w; s < nAct; s += 2) {
        int e = actE[s];
        v8f hc = {};
        const unsigned short* xbase = &xt[m * IN_DIM + hi * 8];
        const unsigned short* bbase = Bh + (size_t)(e * RANK + m) * IN_DIM + hi * 8;
        for (int kb = 0; kb < IN_DIM; kb += 32) {
            FragB a, b;
            uint4 r0 = *(const uint4*)(xbase + kb);
            uint4 r1 = *(const uint4*)(xbase + kb + 16);
            a.u[0] = r0.x; a.u[1] = r0.y; a.u[2] = r0.z; a.u[3] = r0.w;
            a.u[4] = r1.x; a.u[5] = r1.y; a.u[6] = r1.z; a.u[7] = r1.w;
            uint4 q0 = *(const uint4*)(bbase + kb);
            uint4 q1 = *(const uint4*)(bbase + kb + 16);
            b.u[0] = q0.x; b.u[1] = q0.y; b.u[2] = q0.z; b.u[3] = q0.w;
            b.u[4] = q1.x; b.u[5] = q1.y; b.u[6] = q1.z; b.u[7] = q1.w;
            hc = __builtin_amdgcn_wmma_f32_16x16x32_bf16(false, a.v, false, b.v,
                                                         (short)0, hc, false, false);
        }
        for (int j = 0; j < 8; ++j) {
            int row = hi * 8 + j;                       // token row
            float hv = hc[j] * wvs[s * 16 + row];       // fold routing weight
            hbuf[s * 256 + row * 16 + m] = f2bf(hv);    // hbuf[s][token][rank]
        }
    }
    __syncthreads();

    // ---- Stage 2b: Y tiles split across waves; accumulate over experts ----
    const size_t outbase = (size_t)t0 * OUT_DIM;
    for (int nt = w; nt < OUT_DIM / 16; nt += 2) {
        const int ocol = nt * 16 + m;
        v8f yacc = {};
        for (int s = 0; s < nAct; ++s) {
            int e = actE[s];
            FragB a, b;
            // A-operand: rows = tokens from hbuf; K pad slot 16 = routing weight
            uint4 r0 = *(const uint4*)(&hbuf[s * 256 + m * 16 + hi * 8]);
            a.u[0] = r0.x; a.u[1] = r0.y; a.u[2] = r0.z; a.u[3] = r0.w;
            a.u[4] = (hi == 0) ? (unsigned int)f2bf(wvs[s * 16 + m]) : 0u;
            a.u[5] = 0u; a.u[6] = 0u; a.u[7] = 0u;
            // B-operand: Aop[r][o] = A[e][o][r]; K pad slot 16 = bias[e][o]
            uint4 q0 = *(const uint4*)(Ah + (size_t)(e * OUT_DIM + ocol) * RANK + hi * 8);
            b.u[0] = q0.x; b.u[1] = q0.y; b.u[2] = q0.z; b.u[3] = q0.w;
            b.u[4] = (hi == 0) ? (unsigned int)f2bf(bias[(size_t)e * OUT_DIM + ocol]) : 0u;
            b.u[5] = 0u; b.u[6] = 0u; b.u[7] = 0u;
            yacc = __builtin_amdgcn_wmma_f32_16x16x32_bf16(false, a.v, false, b.v,
                                                           (short)0, yacc, false, false);
        }
        for (int j = 0; j < 8; ++j) {
            int row = hi * 8 + j;
            out[outbase + (size_t)row * OUT_DIM + ocol] = yacc[j];
        }
    }
}

// ---------------------------------------------------------------------------
extern "C" void kernel_launch(void* const* d_in, const int* in_sizes, int n_in,
                              void* d_out, int out_size, void* d_ws, size_t ws_size,
                              hipStream_t stream) {
    const float* x    = (const float*)d_in[0];   // (2,4096,1024)
    const float* P    = (const float*)d_in[1];   // (32,1024)
    const float* Bw   = (const float*)d_in[2];   // (32,16,1024)
    const float* Aw   = (const float*)d_in[3];   // (32,1024,16)
    const float* bias = (const float*)d_in[4];   // (32,1024)
    const float* temp = (const float*)d_in[5];   // scalar
    float* out = (float*)d_out;

    const int ntok = in_sizes[0] / IN_DIM;       // 8192

    unsigned short* Bh = (unsigned short*)d_ws;                       // 1 MB bf16
    unsigned short* Ah = Bh + (size_t)N_PROTO * RANK * IN_DIM;        // 1 MB bf16
    float*          p2 = (float*)(Ah + (size_t)N_PROTO * OUT_DIM * RANK);

    prl_convert<<<dim3(512), dim3(256), 0, stream>>>(Bw, Aw, Bh, Ah);
    prl_protosq<<<dim3(1), dim3(32), 0, stream>>>(P, p2);
    prl_main<<<dim3(ntok / 16), dim3(64), 0, stream>>>(x, P, bias, temp, Bh, Ah, p2, out);
}
